// Pointnet2Backbone_50594714747404
// MI455X (gfx1250) — matrix operand order repaired and checked by reading, not compiled, not run on hardware
//
#include <hip/hip_runtime.h>
#include <hip/hip_bf16.h>

// ---------------------------------------------------------------------------
// PointNet++ backbone for gfx1250 (MI455X), wave32, WMMA f16 GEMM MLPs.
// GEMM uses a 2(M)x4(N) register tile strip per wave: 8 v_wmma per K-step,
// 4x A-fragment reuse, 2x B-fragment reuse.
// ---------------------------------------------------------------------------

typedef __attribute__((ext_vector_type(16))) _Float16 v16h;
typedef __attribute__((ext_vector_type(8)))  _Float16 v8h;
typedef __attribute__((ext_vector_type(8)))  float    v8f;

#define NB 2        // batch
#define NPTS 20000  // input points

// ------------------------- small utility kernels ---------------------------

__global__ void split_pc(const float* __restrict__ pc, float* __restrict__ xyz,
                         float* __restrict__ feats, long long total) {
  long long i = (long long)blockIdx.x * blockDim.x + threadIdx.x;
  if (i >= total) return;
  xyz[i * 3 + 0] = pc[i * 6 + 0];
  xyz[i * 3 + 1] = pc[i * 6 + 1];
  xyz[i * 3 + 2] = pc[i * 6 + 2];
  feats[i * 3 + 0] = pc[i * 6 + 3];
  feats[i * 3 + 1] = pc[i * 6 + 4];
  feats[i * 3 + 2] = pc[i * 6 + 5];
}

// fp32 [Nout][K] weights -> f16 [Nout][Kp] (zero-padded K)
__global__ void cvt_w(const float* __restrict__ W, _Float16* __restrict__ Wh,
                      int Nout, int K, int Kp) {
  int i = blockIdx.x * blockDim.x + threadIdx.x;
  if (i >= Nout * Kp) return;
  int o = i / Kp, k = i % Kp;
  Wh[i] = (k < K) ? (_Float16)W[o * K + k] : (_Float16)0.f;
}

__global__ void h2f(const _Float16* __restrict__ s, float* __restrict__ d, long long n) {
  long long i = (long long)blockIdx.x * blockDim.x + threadIdx.x;
  if (i < n) d[i] = (float)s[i];
}

__global__ void copyf(const float* __restrict__ s, float* __restrict__ d, long long n) {
  long long i = (long long)blockIdx.x * blockDim.x + threadIdx.x;
  if (i < n) d[i] = s[i];
}

// ------------------------------ FPS ----------------------------------------
// One workgroup per batch; iterative farthest-point sampling with LDS argmax.
__global__ void fps_kernel(const float* __restrict__ xyz, int N, int npoint,
                           float* __restrict__ dist, int* __restrict__ inds) {
  const int b = blockIdx.x;
  const int tid = threadIdx.x;
  const int T = blockDim.x;  // 256 = 8 wave32
  __shared__ float sMax[256];
  __shared__ int   sArg[256];
  __shared__ float sLast[3];
  const float* X = xyz + (long long)b * N * 3;
  float* D = dist + (long long)b * N;
  for (int i = tid; i < N; i += T) D[i] = 1e10f;
  if (tid == 0) {
    inds[(long long)b * npoint] = 0;
    sLast[0] = X[0]; sLast[1] = X[1]; sLast[2] = X[2];
  }
  __syncthreads();
  for (int t = 1; t < npoint; ++t) {
    const float lx = sLast[0], ly = sLast[1], lz = sLast[2];
    float best = -1.f; int arg = 0;
    for (int i = tid; i < N; i += T) {
      float dx = X[i * 3 + 0] - lx;
      float dy = X[i * 3 + 1] - ly;
      float dz = X[i * 3 + 2] - lz;
      float d = fminf(D[i], dx * dx + dy * dy + dz * dz);
      D[i] = d;
      if (d > best) { best = d; arg = i; }
    }
    sMax[tid] = best; sArg[tid] = arg;
    __syncthreads();
    for (int s = T >> 1; s > 0; s >>= 1) {
      if (tid < s && sMax[tid + s] > sMax[tid]) {
        sMax[tid] = sMax[tid + s]; sArg[tid] = sArg[tid + s];
      }
      __syncthreads();
    }
    if (tid == 0) {
      int a = sArg[0];
      inds[(long long)b * npoint + t] = a;
      sLast[0] = X[a * 3 + 0]; sLast[1] = X[a * 3 + 1]; sLast[2] = X[a * 3 + 2];
    }
    __syncthreads();
  }
}

__global__ void gather3(const float* __restrict__ src, const int* __restrict__ inds,
                        float* __restrict__ dst, int N, int S) {
  long long i = (long long)blockIdx.x * blockDim.x + threadIdx.x;  // over NB*S*3
  long long total = (long long)NB * S * 3;
  if (i >= total) return;
  int c = (int)(i % 3);
  long long bs = i / 3;
  int b = (int)(bs / S);
  int idx = inds[bs];
  dst[i] = src[((long long)b * N + idx) * 3 + c];
}

// ------------------------------ ball query ---------------------------------
// One wave32 per query center; __ballot-based stream compaction of the first
// `nsample` in-radius indices in ascending order (matches top_k(-score)).
__global__ void ball_query_kernel(const float* __restrict__ xyz,
                                  const float* __restrict__ new_xyz,
                                  int N, int S, float r2, int nsample,
                                  int* __restrict__ out, int bS) {
  int gw = (blockIdx.x * blockDim.x + threadIdx.x) >> 5;  // wave32
  int lane = threadIdx.x & 31;
  if (gw >= bS) return;
  const float cx = new_xyz[gw * 3 + 0];
  const float cy = new_xyz[gw * 3 + 1];
  const float cz = new_xyz[gw * 3 + 2];
  const float* X = xyz + (long long)(gw / S) * N * 3;
  int* O = out + (long long)gw * nsample;
  int count = 0, firstIdx = 0, haveFirst = 0;
  for (int base = 0; base < N && count < nsample; base += 32) {
    int i = base + lane;
    bool ok = false;
    if (i < N) {
      float dx = X[i * 3 + 0] - cx;
      float dy = X[i * 3 + 1] - cy;
      float dz = X[i * 3 + 2] - cz;
      ok = (dx * dx + dy * dy + dz * dz) < r2;
    }
    unsigned mask = (unsigned)__ballot(ok);  // wave32: low 32 bits
    if (mask && !haveFirst) { firstIdx = base + __ffs(mask) - 1; haveFirst = 1; }
    int pos = count + __popc(mask & ((1u << lane) - 1u));
    if (ok && pos < nsample) O[pos] = i;
    count += __popc(mask);
  }
  if (count > nsample) count = nsample;
  for (int p = count + lane; p < nsample; p += 32) O[p] = firstIdx;
}

// ------------------------------ grouping -----------------------------------
// Build grouped f16 activation rows: [ (p-center)/r  | feats[idx] | 0-pad ]
__global__ void group_sa(const float* __restrict__ xyz, const float* __restrict__ feats,
                         const float* __restrict__ new_xyz, const int* __restrict__ bidx,
                         int N, int S, int ns, int Cf, int Kp, float invR,
                         _Float16* __restrict__ Xh, long long rows) {
  long long row = (long long)blockIdx.x * blockDim.x + threadIdx.x;
  if (row >= rows) return;
  long long cs = row / ns;  // b*S+s
  int b = (int)(cs / S);
  int idx = bidx[row];
  const float* P = xyz + ((long long)b * N + idx) * 3;
  const float* C = new_xyz + cs * 3;
  _Float16* R = Xh + row * Kp;
  R[0] = (_Float16)((P[0] - C[0]) * invR);
  R[1] = (_Float16)((P[1] - C[1]) * invR);
  R[2] = (_Float16)((P[2] - C[2]) * invR);
  const float* F = feats + ((long long)b * N + idx) * Cf;
  for (int c = 0; c < Cf; ++c) R[3 + c] = (_Float16)F[c];
  for (int k = 3 + Cf; k < Kp; ++k) R[k] = (_Float16)0.f;
}

// ------------------------- WMMA GEMM + scale/bias/relu ----------------------
// Y[M][N] = relu( X[M][Kp] * Wh[N][Kp]^T * scale[N] + bias[N] ), f16 out.
// Fragment layouts (this code's contiguous k order, same for A and B; not AMD's table order):
//  A 16x32: lanes 0-15 hold row l, K0..15; lanes 16-31 K16..31 (contiguous)
//  B 32x16: lanes 0-15 hold col l, K0..15; lanes 16-31 K16..31 (contiguous)
// Register blocking: 2 M-tiles x NSTRIP N-tiles per wave.

__device__ __forceinline__ v16h load_a_frag(const _Float16* p) {
  v8h lo = *(const v8h*)p;         // K base+0..7   (or +16..23)
  v8h hi = *(const v8h*)(p + 8);   // K base+8..15  (or +24..31)
  return __builtin_shufflevector(lo, hi, 0, 1, 2, 3, 4, 5, 6, 7,
                                 8, 9, 10, 11, 12, 13, 14, 15);
}

__device__ __forceinline__ v8f wmma_f16(v16h a, v16h b, v8f c) {
  return __builtin_amdgcn_wmma_f32_16x16x32_f16(
      /*neg_a=*/false, a, /*neg_b=*/false, b,
      /*c_mod=*/(short)0, c, /*reuse_a=*/false, /*reuse_b=*/false);
}

template <int NSTRIP>
__global__ void __launch_bounds__(128)
gemm_bias_relu_wmma(const _Float16* __restrict__ X, int Kp,
                    const _Float16* __restrict__ Wh,
                    const float* __restrict__ scale, const float* __restrict__ bias,
                    _Float16* __restrict__ Y, int Np,
                    int mStrips, int nStrips) {
  int wave = (blockIdx.x * blockDim.x + threadIdx.x) >> 5;
  if (wave >= mStrips * nStrips) return;
  const int ms = wave / nStrips;   // 32-row M strip
  const int nss = wave % nStrips;  // NSTRIP*16-col N strip
  const int lane = threadIdx.x & 31;
  const int half = lane >> 4;      // 0: lanes 0-15, 1: lanes 16-31
  const int l15 = lane & 15;

  const _Float16* arow0 = X + (long long)(ms * 32 + l15) * Kp + (half ? 16 : 0);
  const _Float16* arow1 = arow0 + (long long)16 * Kp;
  const _Float16* brow[NSTRIP];
#pragma unroll
  for (int j = 0; j < NSTRIP; ++j)
    brow[j] = Wh + (long long)((nss * NSTRIP + j) * 16 + l15) * Kp + (half ? 16 : 0);

  v8f acc0[NSTRIP], acc1[NSTRIP];
#pragma unroll
  for (int j = 0; j < NSTRIP; ++j) { acc0[j] = (v8f){}; acc1[j] = (v8f){}; }

  const int kTiles = Kp >> 5;
  for (int kt = 0; kt < kTiles; ++kt) {
    const int ko = kt << 5;
    __builtin_prefetch((const void*)(arow0 + ko + 256), 0, 1);  // global_prefetch_b8
    __builtin_prefetch((const void*)(arow1 + ko + 256), 0, 1);
    v16h a0 = load_a_frag(arow0 + ko);
    v16h a1 = load_a_frag(arow1 + ko);
#pragma unroll
    for (int j = 0; j < NSTRIP; ++j) {
      v16h b = *(const v16h*)(brow[j] + ko);  // 16 contiguous K of one column
      acc0[j] = wmma_f16(a0, b, acc0[j]);
      acc1[j] = wmma_f16(a1, b, acc1[j]);
    }
  }

  // C/D layout: lane<16 -> col=l15 rows 0..7 ; lane>=16 -> col=l15 rows 8..15
#pragma unroll
  for (int j = 0; j < NSTRIP; ++j) {
    const int col = (nss * NSTRIP + j) * 16 + l15;
    const float sc = scale[col];
    const float bi = bias[col];
    long long r0 = (long long)(ms * 32 + half * 8) * Np + col;
    long long r1 = r0 + (long long)16 * Np;
#pragma unroll
    for (int r = 0; r < 8; ++r) {
      float v0 = fmaxf(fmaf(acc0[j][r], sc, bi), 0.f);
      float v1 = fmaxf(fmaf(acc1[j][r], sc, bi), 0.f);
      Y[r0 + (long long)r * Np] = (_Float16)v0;
      Y[r1 + (long long)r * Np] = (_Float16)v1;
    }
  }
}

// --------------------------- max over nsample ------------------------------
__global__ void maxpool_ns(const _Float16* __restrict__ Y, int Np, int ns,
                           float* __restrict__ F, int C, long long total) {
  long long i = (long long)blockIdx.x * blockDim.x + threadIdx.x;
  if (i >= total) return;
  int c = (int)(i % C);
  long long cs = i / C;
  const _Float16* p = Y + cs * (long long)ns * Np + c;
  float m = (float)p[0];
  for (int j = 1; j < ns; ++j) m = fmaxf(m, (float)p[(long long)j * Np]);
  F[i] = m;
}

// ------------------- FP: 3-NN interpolation + concat -----------------------
// One wave per target point (3-NN scan uniform across lanes; lanes split
// the channel work for the interpolated + passthrough features).
__global__ void fp_interp_concat(const float* __restrict__ xyz1, const float* __restrict__ xyz2,
                                 const float* __restrict__ f1, const float* __restrict__ f2,
                                 int n1, int n2, int C1, int C2, int Kp,
                                 _Float16* __restrict__ Xh, int bw) {
  int gw = (blockIdx.x * blockDim.x + threadIdx.x) >> 5;
  int lane = threadIdx.x & 31;
  if (gw >= bw) return;
  int b = gw / n1;
  const float px = xyz1[gw * 3 + 0];
  const float py = xyz1[gw * 3 + 1];
  const float pz = xyz1[gw * 3 + 2];
  const float* X2 = xyz2 + (long long)b * n2 * 3;
  float d0 = 1e30f, d1 = 1e30f, d2 = 1e30f;
  int i0 = 0, i1 = 0, i2 = 0;
  for (int j = 0; j < n2; ++j) {
    float dx = X2[j * 3 + 0] - px;
    float dy = X2[j * 3 + 1] - py;
    float dz = X2[j * 3 + 2] - pz;
    float d = dx * dx + dy * dy + dz * dz;
    if (d < d0)      { d2 = d1; i2 = i1; d1 = d0; i1 = i0; d0 = d; i0 = j; }
    else if (d < d1) { d2 = d1; i2 = i1; d1 = d; i1 = j; }
    else if (d < d2) { d2 = d; i2 = j; }
  }
  float w0 = 1.f / (d0 + 1e-8f), w1 = 1.f / (d1 + 1e-8f), w2 = 1.f / (d2 + 1e-8f);
  float s = w0 + w1 + w2;
  w0 /= s; w1 /= s; w2 /= s;
  const float* F2 = f2 + (long long)b * n2 * C2;
  _Float16* R = Xh + (long long)gw * Kp;
  for (int c = lane; c < C2; c += 32)
    R[c] = (_Float16)(w0 * F2[(long long)i0 * C2 + c] +
                      w1 * F2[(long long)i1 * C2 + c] +
                      w2 * F2[(long long)i2 * C2 + c]);
  const float* F1 = f1 + (long long)gw * C1;
  for (int c = lane; c < C1; c += 32) R[C2 + c] = (_Float16)F1[c];
}

// --------------------------- output assembly -------------------------------
__global__ void transpose_h2f(const _Float16* __restrict__ g, float* __restrict__ out,
                              int n, int C, long long total) {
  long long i = (long long)blockIdx.x * blockDim.x + threadIdx.x;  // (b,c,p)
  if (i >= total) return;
  int p = (int)(i % n);
  long long bc = i / n;
  int c = (int)(bc % C);
  int b = (int)(bc / C);
  out[i] = (float)g[((long long)b * n + p) * C + c];
}

__global__ void transpose_f(const float* __restrict__ g, float* __restrict__ out,
                            int n, int C, long long total) {
  long long i = (long long)blockIdx.x * blockDim.x + threadIdx.x;
  if (i >= total) return;
  int p = (int)(i % n);
  long long bc = i / n;
  int c = (int)(bc % C);
  int b = (int)(bc / C);
  out[i] = g[((long long)b * n + p) * C + c];
}

__global__ void write_inds(const int* __restrict__ inds1, int* __restrict__ outI, int S) {
  int i = blockIdx.x * blockDim.x + threadIdx.x;  // over NB*S
  if (i >= NB * S) return;
  int b = i / S, k = i % S;
  outI[i] = inds1[b * 2048 + k];
}

__global__ void gather_color(const float* __restrict__ pc, const int* __restrict__ inds1,
                             float* __restrict__ out, int S) {
  int i = blockIdx.x * blockDim.x + threadIdx.x;  // over NB*S*3
  if (i >= NB * S * 3) return;
  int c = i % 3;
  int bs = i / 3;
  int b = bs / S, k = bs % S;
  int idx = inds1[b * 2048 + k];
  out[i] = pc[((long long)b * NPTS + idx) * 6 + 3 + c];
}

// ===========================================================================
// host side
// ===========================================================================

struct LD { int K, Kp, N; };

static inline unsigned blocks_for(long long n, int t) {
  return (unsigned)((n + t - 1) / t);
}

extern "C" void kernel_launch(void* const* d_in, const int* in_sizes, int n_in,
                              void* d_out, int out_size, void* d_ws, size_t ws_size,
                              hipStream_t stream) {
  (void)in_sizes; (void)out_size; (void)ws_size;

  // ---- layer dimension tables (K, Kp=ceil32(K), N) ----
  static const LD sa1d[3]  = {{6, 32, 64},    {64, 64, 64},   {64, 64, 128}};
  static const LD sa2d[3]  = {{131, 160, 128},{128, 128, 128},{128, 128, 256}};
  static const LD sa34d[3] = {{259, 288, 128},{128, 128, 128},{128, 128, 256}};
  static const LD fp1d[2]  = {{512, 512, 256},{256, 256, 256}};
  static const LD fp2d[2]  = {{512, 512, 256},{256, 256, 288}};

  // ---- unpack inputs (supports per-leaf or per-list-concatenated layouts) ----
  const float* pc = (const float*)d_in[0];
  const bool leaf = (n_in >= 49);
  int li = 1;
  const float* gbase = nullptr;
  long long goff = 0;
  auto group = [&]() { if (!leaf) { gbase = (const float*)d_in[li++]; goff = 0; } };
  auto take = [&](long long e) -> const float* {
    if (leaf) return (const float*)d_in[li++];
    const float* p = gbase + goff; goff += e; return p;
  };

  const float *W[16], *Sc[16], *Bi[16];
  LD dims[16];
  int nlay = 0;
  auto mod = [&](const LD* d, int nl) {
    group(); for (int i = 0; i < nl; ++i) { W[nlay + i]  = take((long long)d[i].N * d[i].K); }
    group(); for (int i = 0; i < nl; ++i) { Sc[nlay + i] = take(d[i].N); }
    group(); for (int i = 0; i < nl; ++i) { Bi[nlay + i] = take(d[i].N); dims[nlay + i] = d[i]; }
    nlay += nl;
  };
  mod(sa1d, 3); mod(sa2d, 3); mod(sa34d, 3); mod(sa34d, 3); mod(fp1d, 2); mod(fp2d, 2);

  // ---- workspace carve-up (256B aligned) ----
  size_t off = 0;
  auto alloc = [&](size_t bytes) -> void* {
    off = (off + 255) & ~(size_t)255;
    void* p = (char*)d_ws + off;
    off += bytes;
    return p;
  };
  float* xyz0  = (float*)alloc((size_t)NB * NPTS * 3 * 4);
  float* feat0 = (float*)alloc((size_t)NB * NPTS * 3 * 4);
  float* dist  = (float*)alloc((size_t)NB * NPTS * 4);
  int* inds1 = (int*)alloc((size_t)NB * 2048 * 4);
  int* inds2 = (int*)alloc((size_t)NB * 1024 * 4);
  int* inds3 = (int*)alloc((size_t)NB * 512 * 4);
  int* inds4 = (int*)alloc((size_t)NB * 256 * 4);
  float* nx1 = (float*)alloc((size_t)NB * 2048 * 3 * 4);
  float* nx2 = (float*)alloc((size_t)NB * 1024 * 3 * 4);
  float* nx3 = (float*)alloc((size_t)NB * 512 * 3 * 4);
  float* nx4 = (float*)alloc((size_t)NB * 256 * 3 * 4);
  int* bq1 = (int*)alloc((size_t)NB * 2048 * 64 * 4);
  int* bq2 = (int*)alloc((size_t)NB * 1024 * 32 * 4);
  int* bq3 = (int*)alloc((size_t)NB * 512 * 16 * 4);
  int* bq4 = (int*)alloc((size_t)NB * 256 * 16 * 4);
  float* f1 = (float*)alloc((size_t)NB * 2048 * 128 * 4);
  float* f2 = (float*)alloc((size_t)NB * 1024 * 256 * 4);
  float* f3 = (float*)alloc((size_t)NB * 512 * 256 * 4);
  float* f4 = (float*)alloc((size_t)NB * 256 * 256 * 4);
  float* g1f = (float*)alloc((size_t)NB * 512 * 256 * 4);
  _Float16* Wh[16];
  for (int l = 0; l < 16; ++l)
    Wh[l] = (_Float16*)alloc((size_t)dims[l].N * dims[l].Kp * 2);
  _Float16* bufA = (_Float16*)alloc((size_t)17 * 1024 * 1024 * 2);  // 17M halfs
  _Float16* bufB = (_Float16*)alloc((size_t)34 * 1024 * 1024 * 2);  // 34M halfs

  // ---- weight conversion to f16 (K padded to 32) ----
  for (int l = 0; l < 16; ++l) {
    int n = dims[l].N * dims[l].Kp;
    cvt_w<<<blocks_for(n, 256), 256, 0, stream>>>(W[l], Wh[l], dims[l].N, dims[l].K, dims[l].Kp);
  }

  // ---- input split ----
  split_pc<<<blocks_for((long long)NB * NPTS, 256), 256, 0, stream>>>(
      pc, xyz0, feat0, (long long)NB * NPTS);

  // ---- generic WMMA MLP chain (bufA -> bufB -> bufA -> ...) ----
  auto run_chain = [&](long long rows, int lbase, int nl) -> _Float16* {
    _Float16* cur = bufA;
    _Float16* nxt = bufB;
    for (int l = 0; l < nl; ++l) {
      const LD& d = dims[lbase + l];
      int mStrips = (int)(rows / 32);   // rows always a multiple of 32
      int nT = d.N / 16;
      if ((nT & 3) == 0) {
        int nStrips = nT / 4;
        long long waves = (long long)mStrips * nStrips;
        gemm_bias_relu_wmma<4><<<blocks_for(waves * 32, 128), 128, 0, stream>>>(
            cur, d.Kp, Wh[lbase + l], Sc[lbase + l], Bi[lbase + l], nxt, d.N,
            mStrips, nStrips);
      } else {                          // N=288 -> 18 tiles -> strip of 2
        int nStrips = nT / 2;
        long long waves = (long long)mStrips * nStrips;
        gemm_bias_relu_wmma<2><<<blocks_for(waves * 32, 128), 128, 0, stream>>>(
            cur, d.Kp, Wh[lbase + l], Sc[lbase + l], Bi[lbase + l], nxt, d.N,
            mStrips, nStrips);
      }
      _Float16* t = cur; cur = nxt; nxt = t;
    }
    return cur;  // final output buffer
  };

  // ---- one SA stage ----
  auto sa_stage = [&](const float* in_xyz, int Nin, const float* in_feats, int Cf,
                      int npoint, int nsample, float radius,
                      int* inds, float* nxyz, int* bq, int lbase, float* fout) {
    fps_kernel<<<NB, 256, 0, stream>>>(in_xyz, Nin, npoint, dist, inds);
    gather3<<<blocks_for((long long)NB * npoint * 3, 256), 256, 0, stream>>>(
        in_xyz, inds, nxyz, Nin, npoint);
    int bS = NB * npoint;
    ball_query_kernel<<<blocks_for((long long)bS * 32, 256), 256, 0, stream>>>(
        in_xyz, nxyz, Nin, npoint, radius * radius, nsample, bq, bS);
    long long rows = (long long)bS * nsample;
    group_sa<<<blocks_for(rows, 256), 256, 0, stream>>>(
        in_xyz, in_feats, nxyz, bq, Nin, npoint, nsample, Cf,
        dims[lbase].Kp, 1.f / radius, bufA, rows);
    _Float16* y = run_chain(rows, lbase, 3);
    int C = dims[lbase + 2].N;
    long long tot = (long long)bS * C;
    maxpool_ns<<<blocks_for(tot, 256), 256, 0, stream>>>(y, C, nsample, fout, C, tot);
  };

  sa_stage(xyz0, NPTS, feat0, 3,   2048, 64, 0.2f, inds1, nx1, bq1, 0, f1);
  sa_stage(nx1,  2048, f1,   128,  1024, 32, 0.4f, inds2, nx2, bq2, 3, f2);
  sa_stage(nx2,  1024, f2,   256,  512,  16, 0.8f, inds3, nx3, bq3, 6, f3);
  sa_stage(nx3,  512,  f3,   256,  256,  16, 1.2f, inds4, nx4, bq4, 9, f4);

  // ---- FP stages ----
  auto fp_stage = [&](const float* x1, int n1, const float* x2p, int n2,
                      const float* f1p, const float* f2p, int lbase) -> _Float16* {
    int bw = NB * n1;
    fp_interp_concat<<<blocks_for((long long)bw * 32, 256), 256, 0, stream>>>(
        x1, x2p, f1p, f2p, n1, n2, 256, 256, dims[lbase].Kp, bufA, bw);
    return run_chain(bw, lbase, 2);
  };

  _Float16* g1h = fp_stage(nx3, 512, nx4, 256, f3, f4, 12);   // -> bufA (256 ch)
  h2f<<<blocks_for((long long)NB * 512 * 256, 256), 256, 0, stream>>>(
      g1h, g1f, (long long)NB * 512 * 256);
  _Float16* g2h = fp_stage(nx2, 1024, nx3, 512, f2, g1f, 14); // -> bufA (288 ch)

  // ---- output assembly (flat tuple order) ----
  float* out = (float*)d_out;
  const long long o1 = (long long)NB * 288 * 1024;          // fp2_features
  const long long o2 = o1 + (long long)NB * 1024 * 3;       // x2
  const long long o3 = o2 + (long long)NB * 1024;           // fp2_inds (int bits)
  const long long o4 = o3 + (long long)NB * 1024 * 3;       // fp2_color
  const long long o5 = o4 + (long long)NB * 256 * 3;        // x4
  // o5.. : f4 transposed

  transpose_h2f<<<blocks_for(o1, 256), 256, 0, stream>>>(g2h, out, 1024, 288, o1);
  copyf<<<blocks_for((long long)NB * 1024 * 3, 256), 256, 0, stream>>>(
      nx2, out + o1, (long long)NB * 1024 * 3);
  write_inds<<<blocks_for(NB * 1024, 256), 256, 0, stream>>>(
      inds1, (int*)(out + o2), 1024);
  gather_color<<<blocks_for(NB * 1024 * 3, 256), 256, 0, stream>>>(
      pc, inds1, out + o3, 1024);
  copyf<<<blocks_for((long long)NB * 256 * 3, 256), 256, 0, stream>>>(
      nx4, out + o4, (long long)NB * 256 * 3);
  transpose_f<<<blocks_for((long long)NB * 256 * 256, 256), 256, 0, stream>>>(
      f4, out + o5, 256, 256, (long long)NB * 256 * 256);
}
